// HierarchicalPoincareRest_18107582120499
// MI455X (gfx1250) — matrix operand order repaired and checked by reading, not compiled, not run on hardware
//
#include <hip/hip_runtime.h>
#include <math.h>

// ---------------- constants from the reference ----------------
#define C_CURV      1.5f
#define SQRT_C      1.2247448713915890f   // sqrt(1.5)
#define INV_SQRT_C  0.8164965809277260f   // 1/sqrt(1.5)
#define TEMP        0.07f
#define EPS_BALL    1e-5f
// -(1/sqrt(c))/TEMP : folds artanh 2x factor, sign, and temperature
#define OUT_SCALE   (-11.664236870396085f)

// ---------------- vector types ----------------
typedef float    v2f  __attribute__((ext_vector_type(2)));
typedef float    v8f  __attribute__((ext_vector_type(8)));
typedef int      v4i  __attribute__((ext_vector_type(4)));
typedef _Float16 v4h  __attribute__((ext_vector_type(4)));
typedef _Float16 v8h  __attribute__((ext_vector_type(8)));
typedef _Float16 v16h __attribute__((ext_vector_type(16)));

__device__ __forceinline__ v8f wmma_f32(v2f a, v2f b, v8f c) {
  return __builtin_amdgcn_wmma_f32_16x16x4_f32(false, a, false, b, (short)0, c,
                                               false, false);
}
__device__ __forceinline__ v8f wmma_f16(v16h a, v16h b, v8f c) {
  return __builtin_amdgcn_wmma_f32_16x16x32_f16(false, a, false, b, (short)0, c,
                                                false, false);
}

// ---------------- CDNA5 async global->LDS path (guarded) ----------------
#if __has_builtin(__builtin_amdgcn_global_load_async_to_lds_b128) && \
    __has_builtin(__builtin_amdgcn_s_wait_asynccnt)
#define USE_ASYNC_LDS 1
// per the compiler diagnostic: arg0 is int4* in AS(1) (printed "__device__"),
// arg1 is the LDS-side int4* in AS(3)
typedef __attribute__((address_space(1))) v4i glb_v4i;
typedef __attribute__((address_space(3))) v4i lds_v4i;
#else
#define USE_ASYNC_LDS 0
#endif

// ---------------- tiling config ----------------
#define BM 64
#define BN 128
#define KC 32            // f32 GEMM k-chunk
#define KSTR 36          // f32 LDS stride (floats)
#define KCH 32           // f16 GEMM k-chunk (one 16x16x32 step)
#define HSTR 40          // f16 LDS stride (halves): 80B rows, conflict-free
#define NTHREADS 256     // 8 waves (wave32)

// =====================================================================
// MLP GEMM (fp32 WMMA):  C = act(A[M,K] @ W[K,N] + bias)
// =====================================================================
__global__ __launch_bounds__(NTHREADS)
void mlp_gemm_kernel(const float* __restrict__ A, const float* __restrict__ W,
                     const float* __restrict__ bias, float* __restrict__ Cout,
                     int M, int N, int K, int gelu) {
  __shared__ float sA[BM * KSTR];
  __shared__ float sB[BN * KSTR];

  const int tid  = threadIdx.x;
  const int lane = tid & 31;
  const int wave = tid >> 5;
  const int wy   = wave >> 2;
  const int wx   = wave & 3;
  const int m0   = blockIdx.y * BM;
  const int n0   = blockIdx.x * BN;

  v8f acc[2][2];
  for (int i = 0; i < 2; ++i)
    for (int j = 0; j < 2; ++j)
      for (int v = 0; v < 8; ++v) acc[i][j][v] = 0.0f;

  const int klane = (lane < 16) ? 0 : 2;
  const int lrow  = lane & 15;

  for (int k0 = 0; k0 < K; k0 += KC) {
    for (int i = tid; i < (BM * KC / 4); i += NTHREADS) {
      int row = i >> 3;
      int c4  = i & 7;
      const float4 v = *reinterpret_cast<const float4*>(
          A + (size_t)(m0 + row) * K + k0 + c4 * 4);
      *reinterpret_cast<float4*>(&sA[row * KSTR + c4 * 4]) = v;
    }
    for (int i = tid; i < BN * KC; i += NTHREADS) {
      int kk = i >> 7;
      int nn = i & 127;
      sB[nn * KSTR + kk] = W[(size_t)(k0 + kk) * N + n0 + nn];
    }
    __syncthreads();

    for (int kk = 0; kk < KC; kk += 4) {
      v2f a0 = *reinterpret_cast<const v2f*>(&sA[(wy * 32 + lrow) * KSTR + kk + klane]);
      v2f a1 = *reinterpret_cast<const v2f*>(&sA[(wy * 32 + 16 + lrow) * KSTR + kk + klane]);
      v2f b0 = *reinterpret_cast<const v2f*>(&sB[(wx * 32 + lrow) * KSTR + kk + klane]);
      v2f b1 = *reinterpret_cast<const v2f*>(&sB[(wx * 32 + 16 + lrow) * KSTR + kk + klane]);
      acc[0][0] = wmma_f32(a0, b0, acc[0][0]);
      acc[0][1] = wmma_f32(a0, b1, acc[0][1]);
      acc[1][0] = wmma_f32(a1, b0, acc[1][0]);
      acc[1][1] = wmma_f32(a1, b1, acc[1][1]);
    }
    __syncthreads();
  }

  for (int mi = 0; mi < 2; ++mi) {
    for (int ni = 0; ni < 2; ++ni) {
      const int col = n0 + wx * 32 + ni * 16 + (lane & 15);
      const float bv = bias[col];
      const int rowBase = m0 + wy * 32 + mi * 16 + ((lane < 16) ? 0 : 8);
      for (int v = 0; v < 8; ++v) {
        float val = acc[mi][ni][v] + bv;
        if (gelu) val = 0.5f * val * (1.0f + erff(val * 0.70710678118654752f));
        Cout[(size_t)(rowBase + v) * N + col] = val;
      }
    }
  }
}

// =====================================================================
// expmap0 -> split hi/lo f16 + xx = ||hyp||^2  (one wave per row, D=256)
// =====================================================================
__global__ __launch_bounds__(256)
void expmap_split_kernel(const float* __restrict__ f,
                         _Float16* __restrict__ hyp_hi, _Float16* __restrict__ hyp_lo,
                         float* __restrict__ xx, int D) {
  const int wave = threadIdx.x >> 5;
  const int lane = threadIdx.x & 31;
  const int row  = blockIdx.x * 8 + wave;
  const float* fr = f + (size_t)row * D;

  float vals[8];
  float ss = 0.0f;
  for (int i = 0; i < 8; ++i) {
    float v = fr[lane + i * 32];
    vals[i] = v;
    ss += v * v;
  }
  for (int off = 16; off > 0; off >>= 1) ss += __shfl_xor(ss, off, 32);

  const float nrm   = sqrtf(ss);
  const float un    = fmaxf(nrm, EPS_BALL);
  const float t     = tanhf(SQRT_C * un);
  const float scale = t / (SQRT_C * un);
  for (int i = 0; i < 8; ++i) {
    const float    h  = scale * vals[i];
    const _Float16 hh = (_Float16)h;
    const _Float16 hl = (_Float16)(h - (float)hh);
    hyp_hi[(size_t)row * D + lane + i * 32] = hh;
    hyp_lo[(size_t)row * D + lane + i * 32] = hl;
  }
  if (lane == 0) xx[row] = scale * scale * ss;
}

// =====================================================================
// prototypes: yy norms (wave/row) + hi/lo split (elementwise)
// =====================================================================
__global__ __launch_bounds__(256)
void yy_kernel(const float* __restrict__ p, float* __restrict__ yy, int P, int D) {
  const int wave = threadIdx.x >> 5;
  const int lane = threadIdx.x & 31;
  const int row  = blockIdx.x * 8 + wave;
  if (row >= P) return;
  const float* pr = p + (size_t)row * D;
  float ss = 0.0f;
  for (int i = 0; i < 8; ++i) {
    float v = pr[lane + i * 32];
    ss += v * v;
  }
  for (int off = 16; off > 0; off >>= 1) ss += __shfl_xor(ss, off, 32);
  if (lane == 0) yy[row] = ss;
}

__global__ __launch_bounds__(256)
void split_kernel(const float* __restrict__ src, _Float16* __restrict__ hi,
                  _Float16* __restrict__ lo, int n4) {
  const int i = blockIdx.x * 256 + threadIdx.x;
  if (i >= n4) return;
  const float4 v = reinterpret_cast<const float4*>(src)[i];
  v4h h, l;
  h[0] = (_Float16)v.x; l[0] = (_Float16)(v.x - (float)h[0]);
  h[1] = (_Float16)v.y; l[1] = (_Float16)(v.y - (float)h[1]);
  h[2] = (_Float16)v.z; l[2] = (_Float16)(v.z - (float)h[2]);
  h[3] = (_Float16)v.w; l[3] = (_Float16)(v.w - (float)h[3]);
  reinterpret_cast<v4h*>(hi)[i] = h;
  reinterpret_cast<v4h*>(lo)[i] = l;
}

// =====================================================================
// f16 fragment loaders (CDNA5 16x16x32 layouts) from LDS tiles [row][k]
// =====================================================================
__device__ __forceinline__ v16h ldfragA(const _Float16* base, int row, int k0a) {
  union { v16h v; v8h h[2]; } u;
  u.h[0] = *reinterpret_cast<const v8h*>(base + row * HSTR + k0a);
  u.h[1] = *reinterpret_cast<const v8h*>(base + row * HSTR + k0a + 16);
  return u.v;
}
__device__ __forceinline__ v16h ldfragB(const _Float16* base, int row, int kb) {
  union { v16h v; v8h h[2]; } u;
  u.h[0] = *reinterpret_cast<const v8h*>(base + row * HSTR + kb);
  u.h[1] = *reinterpret_cast<const v8h*>(base + row * HSTR + kb + 16);
  return u.v;
}

// =====================================================================
// Distance GEMM, split-f16 (hi*hi + hi*lo + lo*hi), double-buffered LDS
// (async global->LDS when available), fused fast Poincare epilogue.
// =====================================================================
__global__ __launch_bounds__(NTHREADS)
void dist_gemm_f16_kernel(const _Float16* __restrict__ Ahi, const _Float16* __restrict__ Alo,
                          const _Float16* __restrict__ Bhi, const _Float16* __restrict__ Blo,
                          const float* __restrict__ xx, const float* __restrict__ yy,
                          float* __restrict__ out) {
  __shared__ _Float16 sAh[2][BM * HSTR];
  __shared__ _Float16 sAl[2][BM * HSTR];
  __shared__ _Float16 sBh[2][BN * HSTR];
  __shared__ _Float16 sBl[2][BN * HSTR];

  const int tid  = threadIdx.x;
  const int lane = tid & 31;
  const int wave = tid >> 5;
  const int wy   = wave >> 2;
  const int wx   = wave & 3;
  const int m0   = blockIdx.y * BM;
  const int n0   = blockIdx.x * BN;
  const int K    = 256;

  // copy assignment: A = 64 rows x 4 x 16B (one per thread); B = 2x that
  const int crow = tid >> 2;     // 0..63
  const int cc4  = tid & 3;      // 0..3  (8 halves = 16B)

  v8f acc[2][2];
  for (int i = 0; i < 2; ++i)
    for (int j = 0; j < 2; ++j)
      for (int v = 0; v < 8; ++v) acc[i][j][v] = 0.0f;

  const int lrow = lane & 15;
  const int k0a  = (lane < 16) ? 0 : 8;
  const int kb   = (lane < 16) ? 0 : 8;
  const int nch  = K / KCH;   // 8

#if USE_ASYNC_LDS
  auto issue_async = [&](int k0, int buf) {
    const size_t aoff = (size_t)(m0 + crow) * K + k0 + cc4 * 8;
    __builtin_amdgcn_global_load_async_to_lds_b128(
        (glb_v4i*)(Ahi + aoff), (lds_v4i*)&sAh[buf][crow * HSTR + cc4 * 8], 0, 0);
    __builtin_amdgcn_global_load_async_to_lds_b128(
        (glb_v4i*)(Alo + aoff), (lds_v4i*)&sAl[buf][crow * HSTR + cc4 * 8], 0, 0);
#pragma unroll
    for (int j = 0; j < 2; ++j) {
      const size_t boff = (size_t)(n0 + crow + j * 64) * K + k0 + cc4 * 8;
      __builtin_amdgcn_global_load_async_to_lds_b128(
          (glb_v4i*)(Bhi + boff), (lds_v4i*)&sBh[buf][(crow + j * 64) * HSTR + cc4 * 8], 0, 0);
      __builtin_amdgcn_global_load_async_to_lds_b128(
          (glb_v4i*)(Blo + boff), (lds_v4i*)&sBl[buf][(crow + j * 64) * HSTR + cc4 * 8], 0, 0);
    }
  };
#else
  uint4 rAh, rAl, rBh[2], rBl[2];
  auto loadregs = [&](int k0) {
    const size_t aoff = (size_t)(m0 + crow) * K + k0 + cc4 * 8;
    rAh = *reinterpret_cast<const uint4*>(Ahi + aoff);
    rAl = *reinterpret_cast<const uint4*>(Alo + aoff);
#pragma unroll
    for (int j = 0; j < 2; ++j) {
      const size_t boff = (size_t)(n0 + crow + j * 64) * K + k0 + cc4 * 8;
      rBh[j] = *reinterpret_cast<const uint4*>(Bhi + boff);
      rBl[j] = *reinterpret_cast<const uint4*>(Blo + boff);
    }
  };
  auto storelds = [&](int buf) {
    *reinterpret_cast<uint4*>(&sAh[buf][crow * HSTR + cc4 * 8]) = rAh;
    *reinterpret_cast<uint4*>(&sAl[buf][crow * HSTR + cc4 * 8]) = rAl;
#pragma unroll
    for (int j = 0; j < 2; ++j) {
      *reinterpret_cast<uint4*>(&sBh[buf][(crow + j * 64) * HSTR + cc4 * 8]) = rBh[j];
      *reinterpret_cast<uint4*>(&sBl[buf][(crow + j * 64) * HSTR + cc4 * 8]) = rBl[j];
    }
  };
#endif

#if USE_ASYNC_LDS
  issue_async(0, 0);
  int cur = 0;
  for (int c = 0; c < nch; ++c) {
    __builtin_amdgcn_s_wait_asynccnt(0);  // my chunk-c fills landed in LDS
    __syncthreads();                      // everyone's fills landed; prev compute done
    if (c + 1 < nch) issue_async((c + 1) * KCH, cur ^ 1);
#else
  loadregs(0);
  storelds(0);
  int cur = 0;
  for (int c = 0; c < nch; ++c) {
    __syncthreads();
    if (c + 1 < nch) loadregs((c + 1) * KCH);
#endif

    v16h ah0 = ldfragA(&sAh[cur][0], wy * 32 + lrow,      k0a);
    v16h ah1 = ldfragA(&sAh[cur][0], wy * 32 + 16 + lrow, k0a);
    v16h al0 = ldfragA(&sAl[cur][0], wy * 32 + lrow,      k0a);
    v16h al1 = ldfragA(&sAl[cur][0], wy * 32 + 16 + lrow, k0a);
    v16h bh0 = ldfragB(&sBh[cur][0], wx * 32 + lrow,      kb);
    v16h bh1 = ldfragB(&sBh[cur][0], wx * 32 + 16 + lrow, kb);
    v16h bl0 = ldfragB(&sBl[cur][0], wx * 32 + lrow,      kb);
    v16h bl1 = ldfragB(&sBl[cur][0], wx * 32 + 16 + lrow, kb);

    acc[0][0] = wmma_f16(ah0, bh0, acc[0][0]);
    acc[0][1] = wmma_f16(ah0, bh1, acc[0][1]);
    acc[1][0] = wmma_f16(ah1, bh0, acc[1][0]);
    acc[1][1] = wmma_f16(ah1, bh1, acc[1][1]);
    acc[0][0] = wmma_f16(ah0, bl0, acc[0][0]);
    acc[0][1] = wmma_f16(ah0, bl1, acc[0][1]);
    acc[1][0] = wmma_f16(ah1, bl0, acc[1][0]);
    acc[1][1] = wmma_f16(ah1, bl1, acc[1][1]);
    acc[0][0] = wmma_f16(al0, bh0, acc[0][0]);
    acc[0][1] = wmma_f16(al0, bh1, acc[0][1]);
    acc[1][0] = wmma_f16(al1, bh0, acc[1][0]);
    acc[1][1] = wmma_f16(al1, bh1, acc[1][1]);

#if !USE_ASYNC_LDS
    if (c + 1 < nch) storelds(cur ^ 1);
#endif
    cur ^= 1;
  }

  // ---- level-split output mapping (BN=128 divides 1024 & 4096) ----
  size_t obase; int lbase, lsize;
  if (n0 >= 5120)      { obase = (size_t)1024 * 1024 + (size_t)1024 * 4096; lbase = 5120; lsize = 16384; }
  else if (n0 >= 1024) { obase = (size_t)1024 * 1024;                       lbase = 1024; lsize = 4096;  }
  else                 { obase = 0;                                         lbase = 0;    lsize = 1024;  }

  // ---- fast Poincare epilogue: 2*artanh(z) = log((1+z)/(1-z)) ----
  for (int mi = 0; mi < 2; ++mi) {
    const int rowBase = m0 + wy * 32 + mi * 16 + ((lane < 16) ? 0 : 8);
    float xxv[8];
#pragma unroll
    for (int v = 0; v < 8; ++v) xxv[v] = xx[rowBase + v];
    for (int ni = 0; ni < 2; ++ni) {
      const int col   = n0 + wx * 32 + ni * 16 + (lane & 15);
      const float yyc = yy[col];
      const int lcol  = col - lbase;
#pragma unroll
      for (int v = 0; v < 8; ++v) {
        const float xxr = xxv[v];
        const float xy  = acc[mi][ni][v];
        const float a     = 1.0f - 2.0f * C_CURV * xy + C_CURV * yyc;
        const float b     = 1.0f - C_CURV * xxr;
        const float denom = 1.0f - 2.0f * C_CURV * xy + (C_CURV * C_CURV) * xxr * yyc;
        const float num2  = a * a * xxr + b * b * yyc - 2.0f * a * b * xy;
        float z = SQRT_C * __fdividef(sqrtf(fmaxf(num2, 0.0f)), fabsf(denom));
        z = fminf(z, 1.0f - EPS_BALL);             // z >= 0 always
        const float ratio = __fdividef(1.0f + z, 1.0f - z);  // 1-z exact (Sterbenz)
        out[obase + (size_t)(rowBase + v) * lsize + lcol] =
            (OUT_SCALE * 0.5f) * __logf(ratio);
      }
    }
  }
}

// =====================================================================
extern "C" void kernel_launch(void* const* d_in, const int* in_sizes, int n_in,
                              void* d_out, int out_size, void* d_ws, size_t ws_size,
                              hipStream_t stream) {
  const float* x      = (const float*)d_in[0];
  const float* W1     = (const float*)d_in[1];
  const float* b1     = (const float*)d_in[2];
  const float* W2     = (const float*)d_in[3];
  const float* b2     = (const float*)d_in[4];
  const float* W3     = (const float*)d_in[5];
  const float* b3     = (const float*)d_in[6];
  const float* protos = (const float*)d_in[7];

  const int B = 1024, Din = 768, H = 256, D = 256, P = 21504;

  // ---- workspace layout (all sections 16B aligned) ----
  float* ws  = (float*)d_ws;
  float* h1  = ws;                              // 262144 f
  float* h2  = h1 + (size_t)B * H;              // 262144 f
  float* f   = h2 + (size_t)B * H;              // 262144 f
  float* xx  = f + (size_t)B * D;               // 1024 f
  float* yy  = xx + B;                          // 21504 f
  _Float16* hyp_hi = (_Float16*)(yy + P);       // 262144 h
  _Float16* hyp_lo = hyp_hi + (size_t)B * D;    // 262144 h
  _Float16* p_hi   = hyp_lo + (size_t)B * D;    // 5505024 h
  _Float16* p_lo   = p_hi + (size_t)P * D;      // 5505024 h

  const dim3 blk(NTHREADS);
  mlp_gemm_kernel<<<dim3(H / BN, B / BM), blk, 0, stream>>>(x,  W1, b1, h1, B, H, Din, 1);
  mlp_gemm_kernel<<<dim3(H / BN, B / BM), blk, 0, stream>>>(h1, W2, b2, h2, B, H, H,   1);
  mlp_gemm_kernel<<<dim3(D / BN, B / BM), blk, 0, stream>>>(h2, W3, b3, f,  B, D, H,   0);
  expmap_split_kernel<<<B / 8, 256, 0, stream>>>(f, hyp_hi, hyp_lo, xx, D);
  yy_kernel<<<(P + 7) / 8, 256, 0, stream>>>(protos, yy, P, D);
  {
    const int n4 = P * D / 4;
    split_kernel<<<(n4 + 255) / 256, 256, 0, stream>>>(protos, p_hi, p_lo, n4);
  }
  dist_gemm_f16_kernel<<<dim3(P / BN, B / BM), blk, 0, stream>>>(
      hyp_hi, hyp_lo, p_hi, p_lo, xx, yy, (float*)d_out);
}
